// HamiltonianPotential_77704548319437
// MI455X (gfx1250) — matrix operand order repaired and checked by reading, not compiled, not run on hardware
//
#include <hip/hip_runtime.h>
#include <hip/hip_bf16.h>

// ---------------------------------------------------------------------------
// Hamiltonian potential on gfx1250 (MI455X).  B=2, N=256, K=16, A=16, V=32000.
// All 16x16x16 f32 matmuls -> V_WMMA_F32_16X16X4_F32 (4 chained k=4 steps).
// CE logits GEMM (512 x 32000 x 16) -> V_WMMA_F32_16X16X32_F16, online LSE.
// k_align stages per-block-shared operands via GLOBAL_LOAD_ASYNC_TO_LDS_B128
// (ASYNCcnt + s_wait_asynccnt) when the toolchain exposes the builtins.
// ---------------------------------------------------------------------------

typedef float    v8f   __attribute__((ext_vector_type(8)));
typedef float    f32x2 __attribute__((ext_vector_type(2)));
typedef _Float16 v16h  __attribute__((ext_vector_type(16)));
typedef int      v4i   __attribute__((ext_vector_type(4)));

#define B_   2
#define N_   256
#define K_   16
#define A_   16
#define V_   32000
#define BN_  (B_ * N_)
#define EPS_  1e-8f
#define EPS2_ 1e-6f
#define PJ    4          // pairs per block in k_align

#if defined(__has_builtin)
#  if __has_builtin(__builtin_amdgcn_global_load_async_to_lds_b128) && \
      __has_builtin(__builtin_amdgcn_s_wait_asynccnt)
#    define HAVE_ASYNC_LDS 1
#  endif
#endif
#ifndef HAVE_ASYNC_LDS
#  define HAVE_ASYNC_LDS 0
#endif

// same-wave LDS ops are in-order (CDNA5 ISA §7.1); just stop compiler reordering
__device__ inline void lds_fence() {
  asm volatile("" ::: "memory");
  __builtin_amdgcn_wave_barrier();
  asm volatile("" ::: "memory");
}

// ---- WMMA f32 16x16x16 matmul helpers.  Matrices row-major [16][16]. ------
// A-frag (16x4): lane L: M = L%16; vgpr0 K = 2*(L/16), vgpr1 K = 2*(L/16)+1.
// B-frag (4x16): lane L: N = L%16; vgpr0 K = 2*(L/16), vgpr1 K = 2*(L/16)+1.
// C/D (16x16):   vgpr v: M = v + 8*(L/16), N = L%16.
__device__ inline v8f mm16(const float* __restrict__ As,
                           const float* __restrict__ Bs, v8f c) {
  const int lane = threadIdx.x & 31;
  const int half = lane >> 4, l = lane & 15;
#pragma unroll
  for (int kb = 0; kb < 4; ++kb) {
    f32x2 a, b;
    a.x = As[l * 16 + kb * 4 + 2 * half + 0];
    a.y = As[l * 16 + kb * 4 + 2 * half + 1];
    b.x = Bs[(kb * 4 + 2 * half + 0) * 16 + l];
    b.y = Bs[(kb * 4 + 2 * half + 1) * 16 + l];
    c = __builtin_amdgcn_wmma_f32_16x16x4_f32(false, a, false, b,
                                              (short)0, c, false, false);
  }
  return c;
}

// C = A * B^T  (B given row-major; B^T[k][n] = B[n][k])
__device__ inline v8f mm16_bt(const float* __restrict__ As,
                              const float* __restrict__ Bs, v8f c) {
  const int lane = threadIdx.x & 31;
  const int half = lane >> 4, l = lane & 15;
#pragma unroll
  for (int kb = 0; kb < 4; ++kb) {
    f32x2 a, b;
    a.x = As[l * 16 + kb * 4 + 2 * half + 0];
    a.y = As[l * 16 + kb * 4 + 2 * half + 1];
    b.x = Bs[l * 16 + (kb * 4 + 2 * half + 0)];
    b.y = Bs[l * 16 + (kb * 4 + 2 * half + 1)];
    c = __builtin_amdgcn_wmma_f32_16x16x4_f32(false, a, false, b,
                                              (short)0, c, false, false);
  }
  return c;
}

__device__ inline void storeC(float* dst, v8f c) {
  const int lane = threadIdx.x & 31;
  const int half = lane >> 4, l = lane & 15;
#pragma unroll
  for (int v = 0; v < 8; ++v) dst[(v + 8 * half) * 16 + l] = c[v];
}

// ---- wave-level 16x16 Cholesky / solves: lane r (<16) owns row r ----------
// Li[k] = 1/L[k][k] is wave-uniform -> solves use multiplies, no div chains.
__device__ inline void chol16(float (&A)[16], float (&L)[16], float (&Li)[16],
                              float logjit, float* logdet) {
  float ld = 0.f;
#pragma unroll
  for (int k = 0; k < 16; ++k) {
    float akk = __shfl(A[k], k, 32);              // broadcast updated A[k][k]
    float lkk = sqrtf(fmaxf(akk, 1e-30f));
    float linv = 1.0f / lkk;                      // uniform across lanes
    ld += __logf(lkk + logjit);                   // identical on all lanes
    float lrk = A[k] * linv;                      // L[r][k] for r >= k
    L[k] = lrk;
    Li[k] = linv;
#pragma unroll
    for (int c = k + 1; c < 16; ++c)
      A[c] -= lrk * __shfl(lrk, c, 32);           // A[r][c] -= L[r][k]*L[c][k]
  }
  *logdet = 2.f * ld;
}

__device__ inline void fsolve16(const float (&L)[16], const float (&Li)[16],
                                const float (&bb)[16], float (&y)[16]) {
#pragma unroll
  for (int r = 0; r < 16; ++r) {
    float acc = bb[r];
#pragma unroll
    for (int k = 0; k < r; ++k) acc -= __shfl(L[k], r, 32) * y[k];
    y[r] = acc * Li[r];
  }
}

__device__ inline void bsolve16(const float (&L)[16], const float (&Li)[16],
                                const float (&y)[16], float (&z)[16]) {
#pragma unroll
  for (int r = 15; r >= 0; --r) {
    float acc = y[r];
#pragma unroll
    for (int k = r + 1; k < 16; ++k) acc -= __shfl(L[r], k, 32) * z[k];
    z[r] = acc * Li[r];
  }
}

__device__ inline float wave_sum(float v) {
#pragma unroll
  for (int m = 16; m >= 1; m >>= 1) v += __shfl_xor(v, m, 32);
  return v;
}

// ---------------------------------------------------------------------------
__global__ void k_zero(float* out) {
  if (threadIdx.x < B_) out[threadIdx.x] = 0.f;
}

// expm via scaling-and-squaring (s=6, Taylor order 8), all matmuls via WMMA.
__global__ void k_expm(const float* __restrict__ phi,
                       const float* __restrict__ gens,
                       float* __restrict__ expP, float* __restrict__ expM) {
  __shared__ float sM[256], sT[256], sR[256];
  const int bn = blockIdx.x;
  const int lane = threadIdx.x & 31;
  const int half = lane >> 4, l = lane & 15;
  const float* ph = phi + bn * A_;

  float m[8];
#pragma unroll
  for (int v = 0; v < 8; ++v) {
    int row = v + 8 * half;
    float acc = 0.f;
#pragma unroll
    for (int a = 0; a < A_; ++a) acc += ph[a] * gens[a * 256 + row * 16 + l];
    m[v] = acc * (1.0f / 64.0f);                  // pre-scale by 2^-6
  }

  for (int sgn = 0; sgn < 2; ++sgn) {
    float sg = sgn ? -1.f : 1.f;
#pragma unroll
    for (int v = 0; v < 8; ++v) sM[(v + 8 * half) * 16 + l] = sg * m[v];
    lds_fence();

    v8f R;
#pragma unroll
    for (int v = 0; v < 8; ++v) R[v] = ((v + 8 * half) == l) ? 1.f : 0.f;

#pragma unroll
    for (int k = 8; k >= 1; --k) {                // Horner Taylor (constant 1/k)
      storeC(sT, R);
      lds_fence();
      v8f z = {};
      v8f P = mm16(sM, sT, z);
      const float invk = 1.0f / (float)k;
#pragma unroll
      for (int v = 0; v < 8; ++v)
        R[v] = (((v + 8 * half) == l) ? 1.f : 0.f) + P[v] * invk;
    }
    for (int sq = 0; sq < 6; ++sq) {              // square 6 times
      storeC(sR, R);
      lds_fence();
      v8f z = {};
      R = mm16(sR, sR, z);
    }
    float* o = (sgn ? expM : expP) + bn * 256;
#pragma unroll
    for (int v = 0; v < 8; ++v) o[(v + 8 * half) * 16 + l] = R[v];
  }
}

// per (b,n): logdet_q = 2*sum log(diag(chol(Sigma + EPS2 I)) + EPS2)
__global__ void k_ldq(const float* __restrict__ Sigma, float* __restrict__ ldq) {
  const int bn = blockIdx.x;
  const int lane = threadIdx.x & 31, l = lane & 15;
  float A[16], L[16], Li[16], ld;
#pragma unroll
  for (int c = 0; c < 16; ++c)
    A[c] = (lane < 16) ? (Sigma[bn * 256 + l * 16 + c] + ((c == l) ? EPS2_ : 0.f))
                       : ((c == l) ? 1.f : 0.f);
  chol16(A, L, Li, EPS2_, &ld);
  if (lane == 0) ldq[bn] = ld;
}

// V_self: KL(q_n || p_n), one wave per (b,n)
__global__ void k_self(const float* __restrict__ mu, const float* __restrict__ Sigma,
                       const float* __restrict__ mup, const float* __restrict__ Sigp,
                       float* out) {
  const int bn = blockIdx.x, b = bn / N_;
  const int lane = threadIdx.x & 31, l = lane & 15;

  float Ap[16], Aq[16], Aq2[16];
#pragma unroll
  for (int c = 0; c < 16; ++c) {
    if (lane < 16) {
      Ap[c]  = Sigp[bn * 256 + l * 16 + c] + ((c == l) ? EPS_ : 0.f);
      Aq[c]  = Sigma[bn * 256 + l * 16 + c] + ((c == l) ? EPS_ : 0.f);
    } else {
      Ap[c] = (c == l) ? 1.f : 0.f;
      Aq[c] = (c == l) ? 1.f : 0.f;
    }
    Aq2[c] = Aq[c];
  }
  float Lp[16], Lpi[16], Lq[16], Lqi[16], ldp, ldq;
  chol16(Ap,  Lp, Lpi, 0.f, &ldp);                // slogdet(Sp)
  chol16(Aq2, Lq, Lqi, 0.f, &ldq);                // slogdet(Sq)

  float dsc = (lane < 16) ? (mup[bn * 16 + l] - mu[bn * 16 + l]) : 0.f;
  float bb[16];
#pragma unroll
  for (int r = 0; r < 16; ++r) {                  // lane<16: col l of Sq; lane16: dmu
    float d = __shfl(dsc, r, 32);
    bb[r] = (lane < 16) ? Aq[r] : d;
  }
  float y[16], z[16];
  fsolve16(Lp, Lpi, bb, y);
  float ss = 0.f;
#pragma unroll
  for (int r = 0; r < 16; ++r) ss += y[r] * y[r];
  float mah = __shfl(ss, 16, 32);
  bsolve16(Lp, Lpi, y, z);
  float diag = 0.f;
#pragma unroll
  for (int r = 0; r < 16; ++r) diag = (lane == r) ? z[r] : diag;
  float tr = wave_sum(diag);
  float kl = 0.5f * (tr + mah - 16.f + (ldp - ldq));
  if (lane == 0) atomicAdd(&out[b], kl);          // ALPHA = 1
}

// V_align: one wave per (i,j) pair, PJ pairs per block (same b,i).
// E_i and Sigma_i are block-shared -> staged once into LDS (async when possible).
__global__ void k_align(const float* __restrict__ mu, const float* __restrict__ Sigma,
                        const float* __restrict__ beta,
                        const float* __restrict__ expP, const float* __restrict__ expM,
                        const float* __restrict__ ldq, float* out) {
  __shared__ float sOm[PJ][256], sT1[PJ][256], sSt[PJ][256];
  __shared__ float sEi[256], sSi[256];
  __shared__ float sAcc[PJ];
  const int wid = threadIdx.x >> 5;
  const int lane = threadIdx.x & 31, l = lane & 15;

  const int blk = blockIdx.x;                     // B * N * (N/PJ) blocks
  const int b = blk / (N_ * (N_ / PJ));
  const int rem = blk % (N_ * (N_ / PJ));
  const int i = rem / (N_ / PJ);
  const int j = (rem % (N_ / PJ)) * PJ + wid;

  // ---- stage E_i (1KB) + Sigma_i (1KB): 128 threads x 16B = 2KB ----
  {
    const int t = threadIdx.x;
    const float* src = (t < 64) ? (expP + (b * N_ + i) * 256 + (t & 63) * 4)
                                : (Sigma + (b * N_ + i) * 256 + (t & 63) * 4);
    float* dst = (t < 64) ? (sEi + (t & 63) * 4) : (sSi + (t & 63) * 4);
#if HAVE_ASYNC_LDS
    __builtin_amdgcn_global_load_async_to_lds_b128(
        (v4i*)const_cast<float*>(src), (v4i*)dst, 0, 0);
    __builtin_amdgcn_s_wait_asynccnt(0);
#else
    *(float4*)dst = *(const float4*)src;
#endif
  }
  __syncthreads();

  v8f z0 = {};
  v8f Om = mm16(sEi, expM + (b * N_ + j) * 256, z0);            // E_i * F_j
  storeC(&sOm[wid][0], Om);
  lds_fence();
  v8f T1 = mm16(&sOm[wid][0], Sigma + (b * N_ + j) * 256, z0);  // Omega*Sigma_j
  storeC(&sT1[wid][0], T1);
  lds_fence();
  v8f St = mm16_bt(&sT1[wid][0], &sOm[wid][0], z0);             // * Omega^T
  storeC(&sSt[wid][0], St);
  lds_fence();

  float dsc = 0.f;                                // delta = (Omega mu_j - mu_i)[l]
  if (lane < 16) {
    float acc = 0.f;
    const float* mj = mu + (b * N_ + j) * 16;
#pragma unroll
    for (int c = 0; c < 16; ++c) acc += sOm[wid][l * 16 + c] * mj[c];
    dsc = acc - mu[(b * N_ + i) * 16 + l];
  }

  float Ap[16];
#pragma unroll
  for (int c = 0; c < 16; ++c)
    Ap[c] = (lane < 16) ? (sSt[wid][l * 16 + c] + ((c == l) ? EPS2_ : 0.f))
                        : ((c == l) ? 1.f : 0.f);
  float Lp[16], Lpi[16], ldp;
  chol16(Ap, Lp, Lpi, EPS2_, &ldp);

  float bb[16];
#pragma unroll
  for (int r = 0; r < 16; ++r) {                  // lane<16: col l of Sigma_i+eps
    float d = __shfl(dsc, r, 32);
    float sv = (lane < 16) ? (sSi[l * 16 + r] + ((r == l) ? EPS2_ : 0.f)) : 0.f;
    bb[r] = (lane < 16) ? sv : d;
  }
  float y[16], zz[16];
  fsolve16(Lp, Lpi, bb, y);
  float ss = 0.f;
#pragma unroll
  for (int r = 0; r < 16; ++r) ss += y[r] * y[r];
  float mah = __shfl(ss, 16, 32);
  bsolve16(Lp, Lpi, y, zz);
  float diag = 0.f;
#pragma unroll
  for (int r = 0; r < 16; ++r) diag = (lane == r) ? zz[r] : diag;
  float tr = wave_sum(diag);

  float kl = 0.5f * (tr + mah - 16.f + ldp - ldq[b * N_ + i]);
  kl = fmaxf(kl, 0.f);
  float contrib = (i == j) ? 0.f : beta[(b * N_ + i) * N_ + j] * kl;
  if (lane == 0) sAcc[wid] = contrib;
  __syncthreads();
  if (threadIdx.x == 0) {
    float s = 0.f;
#pragma unroll
    for (int w = 0; w < PJ; ++w) s += sAcc[w];
    atomicAdd(&out[b], s);                        // LAMBDA = 1
  }
}

// V_ce: block = (b, 16-row tile); 8 waves stream vocab tiles with f16 WMMA.
__global__ void k_ce(const float* __restrict__ mu, const float* __restrict__ W,
                     const int* __restrict__ tgt, float* out) {
  __shared__ float sm[8][16], ssum[8][16], stg[8][16];
  const int b = blockIdx.x >> 4;
  const int row0 = (blockIdx.x & 15) * 16;
  const int wid = threadIdx.x >> 5;
  const int lane = threadIdx.x & 31;
  const int half = lane >> 4, l = lane & 15;

  // A-frag: mu rows, K padded 16->32 with zeros.
  // 16-bit A 16x32 layout: lane L half h: K = (h%8) + 8*(2*(h/8) + L/16), M = L%16
  v16h a;
#pragma unroll
  for (int h = 0; h < 16; ++h) {
    int kk = (h & 7) + 8 * (2 * (h >> 3) + half);
    float v = (kk < 16) ? mu[(b * N_ + row0 + l) * 16 + kk] : 0.f;
    a[h] = (_Float16)v;
  }
  int tv[8];
#pragma unroll
  for (int v = 0; v < 8; ++v) tv[v] = tgt[b * N_ + row0 + v + 8 * half];

  float m[8], s[8], tl[8];
#pragma unroll
  for (int v = 0; v < 8; ++v) { m[v] = -1e30f; s[v] = 0.f; tl[v] = 0.f; }

  for (int t = wid; t < V_ / 16; t += 8) {
    const int v0 = t * 16;
    // B-frag 32x16: lane L half h: K = (h%8) + 8*(2*(h/8) + L/16), N = L%16; B[K][n] = W[v0+n][K]
    v16h bf;
#pragma unroll
    for (int h = 0; h < 16; ++h) {
      int kk = (h & 7) + 8 * (2 * (h >> 3) + half);
      float v = (kk < 16) ? W[(v0 + l) * 16 + kk] : 0.f;
      bf[h] = (_Float16)v;
    }
    v8f c = {};
    c = __builtin_amdgcn_wmma_f32_16x16x32_f16(false, a, false, bf,
                                               (short)0, c, false, false);
#pragma unroll
    for (int v = 0; v < 8; ++v) {
      float lg = c[v];                            // row v+8*half, col v0+l
      float mn = fmaxf(m[v], lg);
      s[v] = s[v] * __expf(m[v] - mn) + __expf(lg - mn);
      m[v] = mn;
      if (v0 + l == tv[v]) tl[v] = lg;
    }
    __builtin_prefetch(&W[(v0 + 128 + l) * 16], 0, 1);  // next tile for this wave
  }

  // reduce across the 16 column-lanes within each half (masks 8..1 stay in-half)
#pragma unroll
  for (int v = 0; v < 8; ++v) {
#pragma unroll
    for (int msk = 8; msk >= 1; msk >>= 1) {
      float mo = __shfl_xor(m[v], msk, 32);
      float so = __shfl_xor(s[v], msk, 32);
      float mn = fmaxf(m[v], mo);
      s[v] = s[v] * __expf(m[v] - mn) + so * __expf(mo - mn);
      m[v] = mn;
      float to = __shfl_xor(tl[v], msk, 32);
      tl[v] += to;
    }
  }
  if (l == 0) {
#pragma unroll
    for (int v = 0; v < 8; ++v) {
      sm[wid][v + 8 * half] = m[v];
      ssum[wid][v + 8 * half] = s[v];
      stg[wid][v + 8 * half] = tl[v];
    }
  }
  __syncthreads();
  if (threadIdx.x < 16) {
    int r = threadIdx.x;
    float M = sm[0][r], S = ssum[0][r], T = stg[0][r];
#pragma unroll
    for (int w = 1; w < 8; ++w) {
      float mo = sm[w][r], so = ssum[w][r];
      float mn = fmaxf(M, mo);
      S = S * __expf(M - mn) + so * __expf(mo - mn);
      M = mn;
      T += stg[w][r];
    }
    float ce = (M + __logf(S)) - T;               // lse - target logit
#pragma unroll
    for (int msk = 8; msk >= 1; msk >>= 1) ce += __shfl_xor(ce, msk, 32);
    if (r == 0) atomicAdd(&out[b], ce);
  }
}

// ---------------------------------------------------------------------------
extern "C" void kernel_launch(void* const* d_in, const int* in_sizes, int n_in,
                              void* d_out, int out_size, void* d_ws, size_t ws_size,
                              hipStream_t stream) {
  (void)in_sizes; (void)n_in; (void)out_size; (void)ws_size;
  const float* mu     = (const float*)d_in[0];
  const float* Sigma  = (const float*)d_in[1];
  const float* phi    = (const float*)d_in[2];
  const float* mup    = (const float*)d_in[3];
  const float* Sigp   = (const float*)d_in[4];
  const float* beta   = (const float*)d_in[5];
  const float* W_out  = (const float*)d_in[6];
  const float* gens   = (const float*)d_in[7];
  const int*   tgts   = (const int*)d_in[8];
  float* out = (float*)d_out;

  float* expP = (float*)d_ws;                // BN*256
  float* expM = expP + BN_ * 256;            // BN*256
  float* ldq  = expM + BN_ * 256;            // BN

  k_zero<<<1, 32, 0, stream>>>(out);
  k_expm<<<BN_, 32, 0, stream>>>(phi, gens, expP, expM);
  k_ldq<<<BN_, 32, 0, stream>>>(Sigma, ldq);
  k_self<<<BN_, 32, 0, stream>>>(mu, Sigma, mup, Sigp, out);
  k_align<<<B_ * N_ * (N_ / PJ), PJ * 32, 0, stream>>>(mu, Sigma, beta,
                                                       expP, expM, ldq, out);
  k_ce<<<B_ * (N_ / 16), 256, 0, stream>>>(mu, W_out, tgts, out);
}
